// Topdown3DHead_23081154249438
// MI455X (gfx1250) — matrix operand order repaired and checked by reading, not compiled, not run on hardware
//
#include <hip/hip_runtime.h>
#include <hip/hip_bf16.h>

#define KPTS 17
#define HM_H 64
#define HM_W 48
#define HW   (HM_H * HM_W)   // 3072
#define NBATCH 256
#define ROWS (NBATCH * KPTS) // 4352

typedef _Float16 v8h  __attribute__((ext_vector_type(8)));
typedef _Float16 v16h __attribute__((ext_vector_type(16)));
typedef float    v8f  __attribute__((ext_vector_type(8)));

// ---------------------------------------------------------------------------
// Kernel 1: per (n,k) row — heatmap argmax (vectorized b128 scan), posemb
// (6->64), featemb gather, writes keyemb row [320] in f16.
// ---------------------------------------------------------------------------
__global__ __launch_bounds__(128) void head_keyemb_kernel(
    const float* __restrict__ x,        // [N, 256, H, W]
    const float* __restrict__ hm,       // [N, K, H, W]
    const float* __restrict__ bbox,     // [N, 4]
    const float* __restrict__ w_pos,    // [64, 6]
    const float* __restrict__ b_pos,    // [64]
    _Float16* __restrict__ keyemb)      // [ROWS, 320]
{
    const int row = blockIdx.x;         // n*K + k
    const int n   = row / KPTS;
    const int t   = threadIdx.x;

    // Vectorized argmax scan: 3072 floats = 128 threads * 6 float4 each.
    // Rows are 12KB aligned, so float4 loads are 16B aligned.
    const float4* h4 = (const float4*)(hm + (size_t)row * HW);
    float best = -3.402823466e38f;
    int   bidx = 0;
    #pragma unroll
    for (int j = 0; j < 6; ++j) {
        const int i4 = t + 128 * j;
        float4 v = h4[i4];
        const float e[4] = { v.x, v.y, v.z, v.w };
        #pragma unroll
        for (int q = 0; q < 4; ++q) {
            if (e[q] > best) { best = e[q]; bidx = i4 * 4 + q; }  // strict >: first occurrence
        }
    }
    __shared__ float smax[128];
    __shared__ int   sidx[128];
    smax[t] = best; sidx[t] = bidx;
    __syncthreads();
    for (int s = 64; s > 0; s >>= 1) {
        if (t < s) {
            float v2 = smax[t + s]; int i2 = sidx[t + s];
            if (v2 > smax[t] || (v2 == smax[t] && i2 < sidx[t])) {
                smax[t] = v2; sidx[t] = i2;
            }
        }
        __syncthreads();
    }
    const int loc = sidx[0];
    const int xi = loc % HM_W;
    const int yi = loc / HM_W;
    const float kx = (float)xi / (float)HM_W * 2.0f - 1.0f;
    const float ky = (float)yi / (float)HM_H * 2.0f - 1.0f;

    _Float16* out = keyemb + (size_t)row * 320;

    if (t < 64) {
        float in6[6] = { kx, ky, bbox[n * 4 + 0], bbox[n * 4 + 1],
                                 bbox[n * 4 + 2], bbox[n * 4 + 3] };
        float acc = b_pos[t];
        #pragma unroll
        for (int f = 0; f < 6; ++f) acc += w_pos[t * 6 + f] * in6[f];
        out[t] = (_Float16)acc;
    }
    // featemb: 256 channels gathered at (yi, xi); channel stride 12KB (scattered)
    const float* xb = x + (size_t)n * 256 * HW + (size_t)yi * HM_W + xi;
    #pragma unroll
    for (int c = t; c < 256; c += 128) {
        out[64 + c] = (_Float16)xb[(size_t)c * HW];
    }
}

// ---------------------------------------------------------------------------
// Weight conversion kernels (f32 -> f16, plus layout fixups)
// ---------------------------------------------------------------------------
__global__ __launch_bounds__(256) void cvt_f16_kernel(
    const float* __restrict__ src, _Float16* __restrict__ dst, int count)
{
    int i = blockIdx.x * 256 + threadIdx.x;
    if (i < count) dst[i] = (_Float16)src[i];
}

// w_merge [q=256][p=256][k=17] -> dst [q=256][j=4352] with j = k*256 + p
__global__ __launch_bounds__(256) void permute_merge_kernel(
    const float* __restrict__ w, _Float16* __restrict__ dst)
{
    int i = blockIdx.x * 256 + threadIdx.x;   // i = q*4352 + k*256 + p
    if (i >= 256 * 4352) return;
    int q = i / 4352;
    int r = i % 4352;
    int k = r / 256;
    int p = r % 256;
    dst[i] = (_Float16)w[((size_t)q * 256 + p) * KPTS + k];
}

// w_f3 [51,512] -> dst [64,512] zero-padded
__global__ __launch_bounds__(256) void pad_f3_kernel(
    const float* __restrict__ w, _Float16* __restrict__ dst)
{
    int i = blockIdx.x * 256 + threadIdx.x;   // 64*512
    if (i >= 64 * 512) return;
    int o = i / 512;
    dst[i] = (o < 51) ? (_Float16)w[i] : (_Float16)0.0f;
}

// ---------------------------------------------------------------------------
// Generic WMMA GEMM:  Out[M,Nvalid] = A[M,Kd] * B[N32,Kd]^T + bias
// A, B f16 row-major. One wave computes a 16(M) x 32(N) output tile: two
// v_wmma_f32_16x16x32_f16 per K-step sharing one A fragment (halves A
// traffic per FLOP). f32 accumulators; bias added at store.
//
// Register operand layouts used here (wave32, k = 16g + s for A and B):
//   A (16x32 f16): lane m = lane&15, g = lane>>4; two contiguous 16B chunks:
//                  A[m][kb+16g], A[m][kb+16g+8]
//   B (32x16 f16): lane n = lane&15, g = lane>>4; one contiguous 32B chunk:
//                  B[n][kb+16g]
//   C/D (16x16 f32): vgpr r, lane L -> M = r + 8*(L>>4), N = L&15
// ---------------------------------------------------------------------------
__global__ __launch_bounds__(256) void gemm_wmma_kernel(
    const _Float16* __restrict__ A,     // [M, Kd]
    const _Float16* __restrict__ B,     // [N32, Kd]  (rows >= Nvalid readable)
    const float* __restrict__ bias,     // [Nvalid] or nullptr
    void* __restrict__ Out,             // f16 or f32, [M, ldOut]
    int M, int N32, int Kd, int Nvalid, int ldOut, int outF16)
{
    const int wave = threadIdx.x >> 5;
    const int lane = threadIdx.x & 31;
    const int tilesN = N32 >> 5;               // 32-wide N tiles
    const int totalTiles = (M >> 4) * tilesN;
    const int tile = blockIdx.x * 8 + wave;
    if (tile >= totalTiles) return;            // whole-wave exit: EXEC stays all-1s

    const int tM = tile / tilesN;
    const int tN = tile % tilesN;
    const int hr = lane & 15;                  // row within tile (M for A, N for B)
    const int g  = lane >> 4;

    const _Float16* aRow  = A + (size_t)(tM * 16 + hr) * Kd;
    const _Float16* bRow0 = B + (size_t)(tN * 32 + hr) * Kd;
    const _Float16* bRow1 = B + (size_t)(tN * 32 + 16 + hr) * Kd;

    v8f acc0 = {};
    v8f acc1 = {};
    for (int kb = 0; kb < Kd; kb += 32) {
        v8h a0 = *(const v8h*)(aRow + kb + 16 * g);       // 16B aligned
        v8h a1 = *(const v8h*)(aRow + kb + 16 * g + 8);   // 16B aligned
        v16h a;
        #pragma unroll
        for (int i = 0; i < 8; ++i) { a[i] = a0[i]; a[i + 8] = a1[i]; }
        v16h b0 = *(const v16h*)(bRow0 + kb + 16 * g);    // 32B aligned
        v16h b1 = *(const v16h*)(bRow1 + kb + 16 * g);    // 32B aligned
        acc0 = __builtin_amdgcn_wmma_f32_16x16x32_f16(
            false, a, false, b0, (short)0, acc0, false, false);
        acc1 = __builtin_amdgcn_wmma_f32_16x16x32_f16(
            false, a, false, b1, (short)0, acc1, false, false);
    }

    const int nOut0 = tN * 32 + hr;
    const int nOut1 = tN * 32 + 16 + hr;
    const bool nOk0 = (nOut0 < Nvalid);
    const bool nOk1 = (nOut1 < Nvalid);
    const float bv0 = (nOk0 && bias) ? bias[nOut0] : 0.0f;
    const float bv1 = (nOk1 && bias) ? bias[nOut1] : 0.0f;
    #pragma unroll
    for (int r = 0; r < 8; ++r) {
        const int mOut = tM * 16 + r + 8 * g;
        const size_t base = (size_t)mOut * ldOut;
        if (outF16) {
            _Float16* o = (_Float16*)Out;
            if (nOk0) o[base + nOut0] = (_Float16)(acc0[r] + bv0);
            if (nOk1) o[base + nOut1] = (_Float16)(acc1[r] + bv1);
        } else {
            float* o = (float*)Out;
            if (nOk0) o[base + nOut0] = acc0[r] + bv0;
            if (nOk1) o[base + nOut1] = acc1[r] + bv1;
        }
    }
}

// ---------------------------------------------------------------------------
// Launch
// ---------------------------------------------------------------------------
extern "C" void kernel_launch(void* const* d_in, const int* in_sizes, int n_in,
                              void* d_out, int out_size, void* d_ws, size_t ws_size,
                              hipStream_t stream)
{
    const float* x       = (const float*)d_in[0];
    const float* heatmap = (const float*)d_in[1];
    const float* bbox    = (const float*)d_in[2];
    const float* w_pos   = (const float*)d_in[3];
    const float* b_pos   = (const float*)d_in[4];
    const float* w_mlp1  = (const float*)d_in[5];
    const float* b_mlp1  = (const float*)d_in[6];
    const float* w_mlp2  = (const float*)d_in[7];
    const float* b_mlp2  = (const float*)d_in[8];
    const float* w_merge = (const float*)d_in[9];
    const float* b_merge = (const float*)d_in[10];
    const float* w_f1    = (const float*)d_in[11];
    const float* b_f1    = (const float*)d_in[12];
    const float* w_f2    = (const float*)d_in[13];
    const float* b_f2    = (const float*)d_in[14];
    const float* w_f3    = (const float*)d_in[15];
    const float* b_f3    = (const float*)d_in[16];
    float* out = (float*)d_out;

    // Workspace carve-up (all f16, all offsets 64B-aligned)
    char* ws = (char*)d_ws;
    size_t off = 0;
    auto take = [&](size_t halves) {
        _Float16* p = (_Float16*)(ws + off);
        off += halves * sizeof(_Float16);
        off = (off + 63) & ~(size_t)63;
        return p;
    };
    _Float16* keyemb = take((size_t)ROWS * 320);   // 4352 x 320
    _Float16* h1     = take((size_t)ROWS * 128);
    _Float16* h2     = take((size_t)ROWS * 256);
    _Float16* merged = take((size_t)256 * 256);
    _Float16* f1o    = take((size_t)256 * 1024);
    _Float16* f2o    = take((size_t)256 * 512);
    _Float16* wm1    = take((size_t)128 * 320);
    _Float16* wm2    = take((size_t)256 * 128);
    _Float16* wmg    = take((size_t)256 * 4352);
    _Float16* wf1    = take((size_t)1024 * 256);
    _Float16* wf2    = take((size_t)512 * 1024);
    _Float16* wf3    = take((size_t)64 * 512);

    // Stage 0: keyemb (argmax + posemb + gather)
    head_keyemb_kernel<<<ROWS, 128, 0, stream>>>(x, heatmap, bbox, w_pos, b_pos, keyemb);

    // Stage 0b: weight conversions (independent of stage 0)
    cvt_f16_kernel<<<(128 * 320 + 255) / 256, 256, 0, stream>>>(w_mlp1, wm1, 128 * 320);
    cvt_f16_kernel<<<(256 * 128 + 255) / 256, 256, 0, stream>>>(w_mlp2, wm2, 256 * 128);
    cvt_f16_kernel<<<(1024 * 256 + 255) / 256, 256, 0, stream>>>(w_f1, wf1, 1024 * 256);
    cvt_f16_kernel<<<(512 * 1024 + 255) / 256, 256, 0, stream>>>(w_f2, wf2, 512 * 1024);
    permute_merge_kernel<<<(256 * 4352 + 255) / 256, 256, 0, stream>>>(w_merge, wmg);
    pad_f3_kernel<<<(64 * 512 + 255) / 256, 256, 0, stream>>>(w_f3, wf3);

    // Tile math: tiles = (M/16) * (N/32); 8 waves (tiles) per 256-thread block.
    // mlp1: [4352,320] x [128,320]^T -> h1 [4352,128]      tiles=1088 -> 136 blocks
    gemm_wmma_kernel<<<(ROWS / 16) * (128 / 32) / 8, 256, 0, stream>>>(
        keyemb, wm1, b_mlp1, h1, ROWS, 128, 320, 128, 128, 1);
    // mlp2: [4352,128] x [256,128]^T -> h2 [4352,256]      tiles=2176 -> 272 blocks
    gemm_wmma_kernel<<<(ROWS / 16) * (256 / 32) / 8, 256, 0, stream>>>(
        h1, wm2, b_mlp2, h2, ROWS, 256, 128, 256, 256, 1);
    // merge: h2 as [256,4352] x wmg[256,4352]^T -> [256,256]  tiles=128 -> 16 blocks
    gemm_wmma_kernel<<<(256 / 16) * (256 / 32) / 8, 256, 0, stream>>>(
        h2, wmg, b_merge, merged, 256, 256, 4352, 256, 256, 1);
    // f1: [256,256] x [1024,256]^T -> [256,1024]           tiles=512 -> 64 blocks
    gemm_wmma_kernel<<<(256 / 16) * (1024 / 32) / 8, 256, 0, stream>>>(
        merged, wf1, b_f1, f1o, 256, 1024, 256, 1024, 1024, 1);
    // f2: [256,1024] x [512,1024]^T -> [256,512]           tiles=256 -> 32 blocks
    gemm_wmma_kernel<<<(256 / 16) * (512 / 32) / 8, 256, 0, stream>>>(
        f1o, wf2, b_f2, f2o, 256, 512, 1024, 512, 512, 1);
    // f3: [256,512] x [64,512]^T -> d_out f32 [256,51]     tiles=32 -> 4 blocks
    gemm_wmma_kernel<<<(256 / 16) * (64 / 32) / 8, 256, 0, stream>>>(
        f2o, wf3, b_f3, out, 256, 64, 512, 51, 51, 0);
}
